// VallinaRNN_13907104105205
// MI455X (gfx1250) — matrix operand order repaired and checked by reading, not compiled, not run on hardware
//
#include <hip/hip_runtime.h>
#include <hip/hip_bf16.h>

// ---------------------------------------------------------------------------
// VallinaRNN on MI455X (gfx1250):
//   - persistent workgroup per batch element; w[b] resident in LDS as bf16
//     (rows padded to 264 elems -> conflict-free ds_load_b128 A tiles)
//   - matvecs via v_wmma_f32_16x16x32_bf16, vector in column 0 of B; lanes
//     1..15 read a shared zero block (same-address LDS broadcast)
//   - Phase B fused as one K=512 matvec:  t = [Ww | Wu] @ [g ; h]
//   - b and x live in the extraction lanes' registers; vectors ping-pong
//     between two 1 KB bf16 panels -> 3 barriers per step
//   - f32->bf16 via native __bf16 casts -> hardware packed converts (RNE)
// ---------------------------------------------------------------------------

typedef __attribute__((ext_vector_type(16))) __bf16 v16bf;
typedef __attribute__((ext_vector_type(8)))  float  v8f;

#define WPAD 264   // padded row stride of w_bf (528 B = 16 B mod 256 -> no conflicts)
#define PSZ  544   // one panel: 512 K values + pad

union AB16 {
    uint4 q[2];
    v16bf v;
};

// Native conversions: compiler selects HW bf16 converts (RNE) on gfx1250.
__device__ __forceinline__ unsigned short f32_bf16(float f) {
    union { __bf16 h; unsigned short s; } cv;
    cv.h = (__bf16)f;
    return cv.s;
}

__device__ __forceinline__ unsigned int pack2_bf16(float a, float b) {
    union { __bf16 h[2]; unsigned int u; } cv;
    cv.h[0] = (__bf16)a;
    cv.h[1] = (__bf16)b;
    return cv.u;
}

__device__ __forceinline__ uint4 pack_bf16x8(const float* v) {
    return make_uint4(pack2_bf16(v[0], v[1]), pack2_bf16(v[2], v[3]),
                      pack2_bf16(v[4], v[5]), pack2_bf16(v[6], v[7]));
}

// A operand, 16-bit 16x32 layout:
//   lanes 0-15 : row = lane,     K = {0..7} and {16..23} of the chunk
//   lanes 16-31: row = lane-16,  K = {8..15} and {24..31}
__device__ __forceinline__ v16bf aload(const unsigned short* base, int row,
                                       int stride, int kc, int khalf) {
    const unsigned short* p = base + row * stride + kc * 32 + khalf;
    AB16 a;
    a.q[0] = *(const uint4*)(p);
    a.q[1] = *(const uint4*)(p + 16);
    return a.v;
}

// B operand: per-lane base already selects the vector (lanes 0/16) or the
// shared zero block (lanes 1..15, 17..31; same-address broadcast reads).
__device__ __forceinline__ v16bf bload(const unsigned short* base, int kc,
                                       int lane) {
    const unsigned short* p = base + kc * 32 + ((lane & 16) >> 1);
    AB16 b;
    b.q[0] = *(const uint4*)(p);
    b.q[1] = *(const uint4*)(p + 16);
    return b.v;
}

// Prolog: build bf16 weights in d_ws.
//   ws[0 .. 131071]      : combined [Ww | Wu], 256 rows x 512 cols
//   ws[131072 .. 196607] : Wv, 256 x 256
__global__ __launch_bounds__(256)
void rnn_convert_weights(const float* __restrict__ Ww,
                         const float* __restrict__ Wu,
                         const float* __restrict__ Wv,
                         unsigned short* __restrict__ ws) {
    int i = blockIdx.x * 256 + threadIdx.x;          // 0 .. 196607
    if (i < 131072) {
        int r = i >> 9;
        int c = i & 511;
        float v = (c < 256) ? Ww[r * 256 + c] : Wu[r * 256 + (c - 256)];
        ws[i] = f32_bf16(v);
    } else {
        ws[i] = f32_bf16(Wv[i - 131072]);
    }
}

__global__ __launch_bounds__(256)
void rnn_persistent(const float* __restrict__ w,       // [B,256,256]
                    const float* __restrict__ bg,      // [B,256]
                    const unsigned short* __restrict__ wbf,  // [Ww|Wu], Wv bf16
                    float* __restrict__ out,           // [B,256]
                    int nsteps) {
    __shared__ __align__(16) unsigned short w_bf[256 * WPAD];   // 132 KB
    __shared__ __align__(16) unsigned short pz[3 * PSZ];        // P0|P1|Z, 3.2 KB

    const int tid   = threadIdx.x;
    const int wave  = tid >> 5;           // 8 waves (wave32)
    const int lane  = tid & 31;
    const int lrow  = lane & 15;
    const int khalf = (lane < 16) ? 0 : 8;
    const int bidx  = blockIdx.x;
    const int m0 = wave * 16;             // first M tile row base
    const int m1 = (wave + 8) * 16;       // second M tile row base
    const bool extr = ((lane & 15) == 0); // lanes 0 and 16 hold C column 0
    const int rsub = (lane >> 4) << 3;    // lane 0 -> rows +0..7, lane 16 -> +8..15

    const unsigned short* Wab = wbf;            // [Ww | Wu] 256 x 512
    const unsigned short* Wvb = wbf + 131072;   // Wv 256 x 256

    // --- load this batch element's w once, convert f32 -> bf16 into LDS ---
    const float* wsrc = w + (size_t)bidx * 65536;
    for (int i = tid; i < 16384; i += 256) {
        float4 v = ((const float4*)wsrc)[i];
        int r  = i >> 6;
        int c4 = (i & 63) << 2;
        *(uint2*)&w_bf[r * WPAD + c4] =
            make_uint2(pack2_bf16(v.x, v.y), pack2_bf16(v.z, v.w));
    }
    // zero both panels (x0 = h0 = 0) and the shared zero block
    for (int i = tid; i < 3 * PSZ; i += 256) pz[i] = 0;

    // b and x live in the extraction lanes' registers
    float br0[8], br1[8], xr0[8], xr1[8];
    #pragma unroll
    for (int r = 0; r < 8; ++r) { br0[r] = 0.f; br1[r] = 0.f; xr0[r] = 0.f; xr1[r] = 0.f; }
    if (extr) {
        const float* bp = bg + (size_t)bidx * 256;
        #pragma unroll
        for (int r = 0; r < 8; ++r) {
            br0[r] = bp[m0 + rsub + r];
            br1[r] = bp[m1 + rsub + r];
        }
    }
    __syncthreads();

    unsigned short* Pa = pz;              // holds x (K 0..255), h_old (K 256..511 in Pb)
    unsigned short* Pb = pz + PSZ;
    const unsigned short* Z = pz + 2 * PSZ;

    for (int step = 0; step < nsteps; ++step) {
        const unsigned short* ba = extr ? (const unsigned short*)Pa : Z;
        // ---- Phase A: y = w @ x;  g = b - y -> Pb[0..255] ----
        {
            v8f c0 = {}, c1 = {};
            #pragma unroll
            for (int kc = 0; kc < 8; ++kc) {
                v16bf bm = bload(ba, kc, lane);
                v16bf a0 = aload(w_bf, m0 + lrow, WPAD, kc, khalf);
                v16bf a1 = aload(w_bf, m1 + lrow, WPAD, kc, khalf);
                c0 = __builtin_amdgcn_wmma_f32_16x16x32_bf16(
                        false, a0, false, bm, (short)0, c0, false, false);
                c1 = __builtin_amdgcn_wmma_f32_16x16x32_bf16(
                        false, a1, false, bm, (short)0, c1, false, false);
            }
            if (extr) {
                float g0[8], g1[8];
                #pragma unroll
                for (int r = 0; r < 8; ++r) { g0[r] = br0[r] - c0[r]; g1[r] = br1[r] - c1[r]; }
                *(uint4*)&Pb[m0 + rsub] = pack_bf16x8(g0);
                *(uint4*)&Pb[m1 + rsub] = pack_bf16x8(g1);
            }
        }
        __syncthreads();

        // ---- Phase B: t = [Ww | Wu] @ [g ; h_old]  (K = 512, from Pb);
        //      h = relu(t) -> Pa[0..255] (for C) and Pa[256..511] (next step) ----
        {
            const unsigned short* bb = extr ? (const unsigned short*)Pb : Z;
            v8f c0 = {}, c1 = {};
            #pragma unroll
            for (int kc = 0; kc < 16; ++kc) {
                v16bf bm = bload(bb, kc, lane);
                v16bf a0 = aload(Wab, m0 + lrow, 512, kc, khalf);
                v16bf a1 = aload(Wab, m1 + lrow, 512, kc, khalf);
                c0 = __builtin_amdgcn_wmma_f32_16x16x32_bf16(
                        false, a0, false, bm, (short)0, c0, false, false);
                c1 = __builtin_amdgcn_wmma_f32_16x16x32_bf16(
                        false, a1, false, bm, (short)0, c1, false, false);
            }
            if (extr) {
                float h0[8], h1[8];
                #pragma unroll
                for (int r = 0; r < 8; ++r) {
                    h0[r] = c0[r] > 0.f ? c0[r] : 0.f;
                    h1[r] = c1[r] > 0.f ? c1[r] : 0.f;
                }
                uint4 q0 = pack_bf16x8(h0);
                uint4 q1 = pack_bf16x8(h1);
                *(uint4*)&Pa[m0 + rsub]       = q0;
                *(uint4*)&Pa[m1 + rsub]       = q1;
                *(uint4*)&Pa[256 + m0 + rsub] = q0;
                *(uint4*)&Pa[256 + m1 + rsub] = q1;
            }
        }
        __syncthreads();

        // ---- Phase C: x += Wv @ h  (h from Pa); x_new -> Pb[0..255] ----
        {
            v8f c0 = {}, c1 = {};
            #pragma unroll
            for (int kc = 0; kc < 8; ++kc) {
                v16bf bm = bload(ba, kc, lane);   // same base as Phase A (Pa)
                v16bf a0 = aload(Wvb, m0 + lrow, 256, kc, khalf);
                v16bf a1 = aload(Wvb, m1 + lrow, 256, kc, khalf);
                c0 = __builtin_amdgcn_wmma_f32_16x16x32_bf16(
                        false, a0, false, bm, (short)0, c0, false, false);
                c1 = __builtin_amdgcn_wmma_f32_16x16x32_bf16(
                        false, a1, false, bm, (short)0, c1, false, false);
            }
            if (extr) {
                #pragma unroll
                for (int r = 0; r < 8; ++r) { xr0[r] += c0[r]; xr1[r] += c1[r]; }
                *(uint4*)&Pb[m0 + rsub] = pack_bf16x8(xr0);
                *(uint4*)&Pb[m1 + rsub] = pack_bf16x8(xr1);
            }
        }
        __syncthreads();

        // swap panels: Pb now holds x (K 0..255); h sits in Pa (K 256..511)
        unsigned short* t = Pa; Pa = Pb; Pb = t;
    }

    if (extr) {
        float* op = out + (size_t)bidx * 256;
        #pragma unroll
        for (int r = 0; r < 8; ++r) {
            op[m0 + rsub + r] = xr0[r];
            op[m1 + rsub + r] = xr1[r];
        }
    }
}

extern "C" void kernel_launch(void* const* d_in, const int* in_sizes, int n_in,
                              void* d_out, int out_size, void* d_ws, size_t ws_size,
                              hipStream_t stream) {
    const float* w  = (const float*)d_in[0];   // [B,256,256]
    const float* b  = (const float*)d_in[1];   // [B,256]
    const float* Ww = (const float*)d_in[2];   // [256,256]
    const float* Wu = (const float*)d_in[3];   // [256,256]
    const float* Wv = (const float*)d_in[4];   // [256,256]
    float* out = (float*)d_out;
    unsigned short* wsbf = (unsigned short*)d_ws;  // needs >= 384 KiB

    const int B = in_sizes[1] / 256;           // 2048
    const int K = 20;

    rnn_convert_weights<<<196608 / 256, 256, 0, stream>>>(Ww, Wu, Wv, wsbf);
    rnn_persistent<<<B, 256, 0, stream>>>(w, b, wsbf, out, K);
}
